// SE3Transformer_34308198761148
// MI455X (gfx1250) — matrix operand order repaired and checked by reading, not compiled, not run on hardware
//
#include <hip/hip_runtime.h>
#include <hip/hip_bf16.h>
#include <math.h>

typedef __attribute__((ext_vector_type(16))) _Float16 v16h;
typedef __attribute__((ext_vector_type(8)))  float    v8f;

#define EPSV 1e-6f

// ---------- ordered-float encoding for atomicMax-based segment max ----------
__device__ __forceinline__ unsigned fenc(float f) {
    unsigned u = __float_as_uint(f);
    return (u & 0x80000000u) ? ~u : (u | 0x80000000u);
}
__device__ __forceinline__ float fdec(unsigned u) {
    unsigned v = (u & 0x80000000u) ? (u & 0x7FFFFFFFu) : ~u;
    return __uint_as_float(v);
}

// ---------- per-layer compile-time config ----------
template<int L> struct LC;
// (m0_in, m1_in, ck=2) ; sizes order: 00,10,01,11
template<> struct LC<0> { enum { M0=0, M1=8, TOT=72,  ROUT=144, NT=9,  HAS00=0, HAS10=1, HAS01=0,
                                 NEEDQ0=0, OUT0=1, M0O=8, M1O=8, NORM=1, DK=6, STRIDE=NT*16+1 }; };
template<> struct LC<1> { enum { M0=8, M1=8, TOT=104, ROUT=208, NT=13, HAS00=1, HAS10=1, HAS01=1,
                                 NEEDQ0=1, OUT0=1, M0O=8, M1O=8, NORM=1, DK=8, STRIDE=NT*16+1 }; };
template<> struct LC<2> { enum { M0=8, M1=8, TOT=70,  ROUT=140, NT=9,  HAS00=0, HAS10=0, HAS01=1,
                                 NEEDQ0=0, OUT0=0, M0O=0, M1O=2, NORM=0, DK=6, STRIDE=NT*16+1 }; };

// ---------- edge geometry ----------
__global__ void geom_kernel(const float* __restrict__ x, const int* __restrict__ src,
                            const int* __restrict__ dst, float* __restrict__ rhat,
                            float* __restrict__ distb, int E) {
    int e = blockIdx.x * blockDim.x + threadIdx.x;
    if (e >= E) return;
    int s = src[e], d = dst[e];
    float r0 = x[d*3+0] - x[s*3+0];
    float r1 = x[d*3+1] - x[s*3+1];
    float r2 = x[d*3+2] - x[s*3+2];
    float dd = sqrtf(r0*r0 + r1*r1 + r2*r2 + 1e-12f);
    distb[e] = dd;
    float inv = 1.0f / dd;
    rhat[(long long)e*3+0] = r0*inv;
    rhat[(long long)e*3+1] = r1*inv;
    rhat[(long long)e*3+2] = r2*inv;
}

// ---------- per-node init of softmax / aggregation state ----------
__global__ void node_init_kernel(unsigned* __restrict__ mbuf, float* __restrict__ sbuf,
                                 float* __restrict__ agg0, float* __restrict__ agg1, int N) {
    int n = blockIdx.x * blockDim.x + threadIdx.x;
    if (n >= N) return;
    mbuf[n] = fenc(-3.0e38f);
    sbuf[n] = 0.f;
    agg0[n*2+0] = 0.f; agg0[n*2+1] = 0.f;
    #pragma unroll
    for (int k = 0; k < 6; ++k) agg1[(long long)n*6+k] = 0.f;
}

// ---------- per-node queries ----------
template<int L>
__global__ void node_q_kernel(const float* __restrict__ h1, const float* __restrict__ h0,
                              const float* __restrict__ wq1, const float* __restrict__ wq0,
                              float* __restrict__ q1, float* __restrict__ q0, int N) {
    enum { M0 = LC<L>::M0, M1 = LC<L>::M1 };
    int n = blockIdx.x * blockDim.x + threadIdx.x;
    if (n >= N) return;
    #pragma unroll
    for (int o = 0; o < 2; ++o) {
        float a0 = 0.f, a1 = 0.f, a2 = 0.f;
        #pragma unroll
        for (int i = 0; i < M1; ++i) {
            float w = wq1[o*M1 + i];
            a0 += w * h1[((long long)n*M1 + i)*3 + 0];
            a1 += w * h1[((long long)n*M1 + i)*3 + 1];
            a2 += w * h1[((long long)n*M1 + i)*3 + 2];
        }
        q1[((long long)n*2 + o)*3 + 0] = a0;
        q1[((long long)n*2 + o)*3 + 1] = a1;
        q1[((long long)n*2 + o)*3 + 2] = a2;
    }
    if (LC<L>::NEEDQ0) {
        #pragma unroll
        for (int o = 0; o < 2; ++o) {
            float acc = 0.f;
            #pragma unroll
            for (int i = 0; i < M0; ++i) acc += wq0[o*M0 + i] * h0[(long long)n*M0 + i];
            q0[n*2 + o] = acc;
        }
    }
}

// ---------- fused radial-MLP (WMMA) + TP messages + logits ----------
template<int L>
__global__ __launch_bounds__(128) void edge_wmma_kernel(
    const float* __restrict__ distb, const float* __restrict__ efeat,
    const float* __restrict__ rhat,
    const int* __restrict__ src, const int* __restrict__ dst,
    const float* __restrict__ h1cur, const float* __restrict__ h0cur,
    const float* __restrict__ rw1, const float* __restrict__ rb1,
    const float* __restrict__ rw2, const float* __restrict__ rb2,
    const float* __restrict__ q1, const float* __restrict__ q0,
    float* __restrict__ logits, float* __restrict__ v0w, float* __restrict__ v1w,
    unsigned* __restrict__ mbuf, int E)
{
    enum { M0 = LC<L>::M0, M1 = LC<L>::M1, TOT = LC<L>::TOT, ROUT = LC<L>::ROUT,
           NT = LC<L>::NT, STRIDE = LC<L>::STRIDE };
    constexpr int M1A = M1 + 1;
    __shared__ float Rld[4][16][STRIDE];

    const int w    = threadIdx.x >> 5;
    const int lane = threadIdx.x & 31;
    const int half = lane >> 4;
    const int m    = lane & 15;
    const long long e = ((long long)blockIdx.x * 4 + w) * 16 + m;
    const bool ev = (e < E);

    // --- Stage 1 (K=2 layer) computed straight into the WMMA A fragment ---
    float f0 = ev ? distb[e] : 0.f;
    float f1 = ev ? efeat[e] : 0.f;
    v16h afrag;
    #pragma unroll
    for (int j = 0; j < 8; ++j) {
        int k  = 8*half + j;            // A layout: lanes0-15 K0..7 / lanes16-31 K8..15
        int k2 = 16 + 8*half + j;       //           then K16..23 / K24..31
        float h  = rb1[k]  + f0*rw1[k]  + f1*rw1[32+k];
        float h2 = rb1[k2] + f0*rw1[k2] + f1*rw1[32+k2];
        afrag[j]   = (_Float16)fmaxf(h,  0.f);
        afrag[8+j] = (_Float16)fmaxf(h2, 0.f);
    }

    // --- Stage 2: R = hidden @ rw2 + rb2 via v_wmma_f32_16x16x32_f16 ---
    for (int nt = 0; nt < NT; ++nt) {
        int c = nt*16 + m;              // this lane's output column
        bool cv = (c < ROUT);
        v16h bfrag;
        #pragma unroll
        for (int j = 0; j < 16; ++j) {  // B layout: lanes0-15 K0..7,16..23, lanes16-31 K8..15,24..31
            int k = (j < 8) ? (8*half + j) : (16 + 8*half + (j - 8));
            bfrag[j] = (_Float16)(cv ? rw2[(long long)k*ROUT + c] : 0.f);
        }
        v8f dacc = {0.f,0.f,0.f,0.f,0.f,0.f,0.f,0.f};
        dacc = __builtin_amdgcn_wmma_f32_16x16x32_f16(false, afrag, false, bfrag,
                                                      (short)0, dacc, false, false);
        float bias = cv ? rb2[c] : 0.f;
        #pragma unroll
        for (int r = 0; r < 8; ++r) {   // D layout: row = r + 8*half, col = c
            if (cv) Rld[w][r + 8*half][c] = dacc[r] + bias;
        }
    }
    __syncthreads();
    if (!ev) return;

    // --- per-edge TP message math: half==0 -> K part, half==1 -> V part ---
    const int sN = src[e], dN = dst[e];
    const float rh0 = rhat[(long long)e*3+0];
    const float rh1 = rhat[(long long)e*3+1];
    const float rh2 = rhat[(long long)e*3+2];
    float h1s[M1A][3];
    #pragma unroll
    for (int i = 0; i < M1; ++i) {
        h1s[i][0] = h1cur[((long long)sN*M1 + i)*3 + 0];
        h1s[i][1] = h1cur[((long long)sN*M1 + i)*3 + 1];
        h1s[i][2] = h1cur[((long long)sN*M1 + i)*3 + 2];
    }
    h1s[M1][0] = rh0; h1s[M1][1] = rh1; h1s[M1][2] = rh2;
    float inner[M1A];
    #pragma unroll
    for (int i = 0; i < M1A; ++i)
        inner[i] = rh0*h1s[i][0] + rh1*h1s[i][1] + rh2*h1s[i][2];
    float h0s[(M0 > 0) ? M0 : 1];
    if (M0 > 0) {
        #pragma unroll
        for (int i = 0; i < M0; ++i) h0s[i] = h0cur[(long long)sN*M0 + i];
    }

    const float* Rr = &Rld[w][m][0];
    int off = (half == 0) ? 0 : TOT;
    float m0o0 = 0.f, m0o1 = 0.f;
    float m1o[2][3] = {{0.f,0.f,0.f},{0.f,0.f,0.f}};

    if (LC<L>::HAS00) {                       // '00': ck x m0
        #pragma unroll
        for (int i = 0; i < M0; ++i) {
            m0o0 += Rr[off + i]      * h0s[i];
            m0o1 += Rr[off + M0 + i] * h0s[i];
        }
        off += 2*M0;
    }
    if (LC<L>::HAS10) {                       // '10': ck x m1a on inner
        #pragma unroll
        for (int i = 0; i < M1A; ++i) {
            m0o0 += Rr[off + i]        * inner[i];
            m0o1 += Rr[off + M1A + i]  * inner[i];
        }
        off += 2*M1A;
    }
    if (LC<L>::HAS01) {                       // '01': (W h0s) outer rhat
        float sa = 0.f, sb = 0.f;
        #pragma unroll
        for (int i = 0; i < M0; ++i) {
            sa += Rr[off + i]      * h0s[i];
            sb += Rr[off + M0 + i] * h0s[i];
        }
        m1o[0][0] += sa*rh0; m1o[0][1] += sa*rh1; m1o[0][2] += sa*rh2;
        m1o[1][0] += sb*rh0; m1o[1][1] += sb*rh1; m1o[1][2] += sb*rh2;
        off += 2*M0;
    }
    // '11': 3 x ck x m1a  (identity, cross, symmetric-traceless)
    #pragma unroll
    for (int i = 0; i < M1A; ++i) {
        float cr0 = rh1*h1s[i][2] - rh2*h1s[i][1];
        float cr1 = rh2*h1s[i][0] - rh0*h1s[i][2];
        float cr2 = rh0*h1s[i][1] - rh1*h1s[i][0];
        float sy0 = inner[i]*rh0 - h1s[i][0]*(1.f/3.f);
        float sy1 = inner[i]*rh1 - h1s[i][1]*(1.f/3.f);
        float sy2 = inner[i]*rh2 - h1s[i][2]*(1.f/3.f);
        #pragma unroll
        for (int o = 0; o < 2; ++o) {
            float wid = Rr[off +            o*M1A + i];
            float wcr = Rr[off + 2*M1A +    o*M1A + i];
            float wsy = Rr[off + 4*M1A +    o*M1A + i];
            m1o[o][0] += wid*h1s[i][0] + wcr*cr0 + wsy*sy0;
            m1o[o][1] += wid*h1s[i][1] + wcr*cr1 + wsy*sy1;
            m1o[o][2] += wid*h1s[i][2] + wcr*cr2 + wsy*sy2;
        }
    }

    if (half == 0) {  // K part -> logit + segment max
        float lg = 0.f;
        #pragma unroll
        for (int o = 0; o < 2; ++o)
            #pragma unroll
            for (int d = 0; d < 3; ++d)
                lg += m1o[o][d] * q1[((long long)dN*2 + o)*3 + d];
        if (LC<L>::NEEDQ0)
            lg += m0o0*q0[dN*2+0] + m0o1*q0[dN*2+1];
        lg *= rsqrtf((float)LC<L>::DK);
        logits[e] = lg;
        atomicMax(&mbuf[dN], fenc(lg));
    } else {          // V part -> store
        #pragma unroll
        for (int o = 0; o < 2; ++o) {
            v1w[((long long)e*2 + o)*3 + 0] = m1o[o][0];
            v1w[((long long)e*2 + o)*3 + 1] = m1o[o][1];
            v1w[((long long)e*2 + o)*3 + 2] = m1o[o][2];
        }
        if (LC<L>::OUT0) { v0w[(long long)e*2+0] = m0o0; v0w[(long long)e*2+1] = m0o1; }
    }
}

// ---------- softmax exp + segment sum ----------
__global__ void edge_exp_kernel(const float* __restrict__ logits, const int* __restrict__ dst,
                                const unsigned* __restrict__ mbuf, float* __restrict__ sbuf,
                                float* __restrict__ ew, int E) {
    int e = blockIdx.x * blockDim.x + threadIdx.x;
    if (e >= E) return;
    int d = dst[e];
    float v = expf(logits[e] - fdec(mbuf[d]));
    ew[e] = v;
    atomicAdd(&sbuf[d], v);
}

// ---------- alpha-weighted aggregation ----------
template<int L>
__global__ void edge_agg_kernel(const float* __restrict__ ew, const float* __restrict__ sbuf,
                                const int* __restrict__ dst,
                                const float* __restrict__ v0w, const float* __restrict__ v1w,
                                float* __restrict__ agg0, float* __restrict__ agg1, int E) {
    int e = blockIdx.x * blockDim.x + threadIdx.x;
    if (e >= E) return;
    int d = dst[e];
    float alpha = ew[e] / (sbuf[d] + EPSV);
    #pragma unroll
    for (int k = 0; k < 6; ++k)
        atomicAdd(&agg1[(long long)d*6 + k], alpha * v1w[(long long)e*6 + k]);
    if (LC<L>::OUT0) {
        atomicAdd(&agg0[d*2+0], alpha * v0w[(long long)e*2+0]);
        atomicAdd(&agg0[d*2+1], alpha * v0w[(long long)e*2+1]);
    }
}

// ---------- output projection + gated norm ----------
template<int L>
__global__ void node_out_kernel(const float* __restrict__ agg0, const float* __restrict__ agg1,
                                const float* __restrict__ h0cur, const float* __restrict__ h1cur,
                                const float* __restrict__ wo0, const float* __restrict__ wo1,
                                const float* __restrict__ s0, const float* __restrict__ b0,
                                const float* __restrict__ s1, const float* __restrict__ b1,
                                float* __restrict__ h0out, float* __restrict__ h1out, int N) {
    enum { M0 = LC<L>::M0, M1 = LC<L>::M1, M0O = LC<L>::M0O, M1O = LC<L>::M1O };
    int n = blockIdx.x * blockDim.x + threadIdx.x;
    if (n >= N) return;
    #pragma unroll
    for (int o = 0; o < M1O; ++o) {
        float o0 = 0.f, o1 = 0.f, o2 = 0.f;
        #pragma unroll
        for (int i = 0; i < 2; ++i) {
            float wv = wo1[o*(2+M1) + i];
            o0 += wv * agg1[(long long)n*6 + i*3 + 0];
            o1 += wv * agg1[(long long)n*6 + i*3 + 1];
            o2 += wv * agg1[(long long)n*6 + i*3 + 2];
        }
        #pragma unroll
        for (int i = 0; i < M1; ++i) {
            float wv = wo1[o*(2+M1) + 2 + i];
            o0 += wv * h1cur[((long long)n*M1 + i)*3 + 0];
            o1 += wv * h1cur[((long long)n*M1 + i)*3 + 1];
            o2 += wv * h1cur[((long long)n*M1 + i)*3 + 2];
        }
        if (LC<L>::NORM) {
            float nn = sqrtf(o0*o0 + o1*o1 + o2*o2 + 1e-12f);
            float g  = fmaxf(nn*s1[o] + b1[o], 0.f) / (nn + EPSV);
            o0 *= g; o1 *= g; o2 *= g;
        }
        h1out[((long long)n*M1O + o)*3 + 0] = o0;
        h1out[((long long)n*M1O + o)*3 + 1] = o1;
        h1out[((long long)n*M1O + o)*3 + 2] = o2;
    }
    if (LC<L>::OUT0) {
        #pragma unroll
        for (int o = 0; o < M0O; ++o) {
            float acc = 0.f;
            #pragma unroll
            for (int i = 0; i < 2; ++i) acc += wo0[o*(2+M0) + i] * agg0[n*2 + i];
            #pragma unroll
            for (int i = 0; i < M0; ++i) acc += wo0[o*(2+M0) + 2 + i] * h0cur[(long long)n*M0 + i];
            if (LC<L>::NORM) {
                float na = fabsf(acc);
                acc = fmaxf(na*s0[o] + b0[o], 0.f) * acc / (na + EPSV);
            }
            h0out[(long long)n*M0O + o] = acc;
        }
    }
}

// ---------- host-side per-layer driver ----------
template<int L>
static void run_layer(const float* h1cur, const float* h0cur,
                      const float* rw1, const float* rb1, const float* rw2, const float* rb2,
                      const float* wq1, const float* wq0, const float* wo1, const float* wo0,
                      const float* s0, const float* b0, const float* s1, const float* b1,
                      float* h1out, float* h0out,
                      const float* distb, const float* efeat, const float* rhat,
                      const int* src, const int* dst,
                      float* q1, float* q0, unsigned* mbuf, float* sbuf,
                      float* agg0, float* agg1,
                      float* logits, float* ew, float* v0w, float* v1w,
                      int N, int E, hipStream_t stream) {
    int nb = (N + 255) / 256;
    int eb = (E + 255) / 256;
    int tb = (E + 63) / 64;  // 4 waves/block, 16 edges/wave
    node_q_kernel<L><<<nb, 256, 0, stream>>>(h1cur, h0cur, wq1, wq0, q1, q0, N);
    node_init_kernel<<<nb, 256, 0, stream>>>(mbuf, sbuf, agg0, agg1, N);
    edge_wmma_kernel<L><<<tb, 128, 0, stream>>>(distb, efeat, rhat, src, dst, h1cur, h0cur,
                                                rw1, rb1, rw2, rb2, q1, q0,
                                                logits, v0w, v1w, mbuf, E);
    edge_exp_kernel<<<eb, 256, 0, stream>>>(logits, dst, mbuf, sbuf, ew, E);
    edge_agg_kernel<L><<<eb, 256, 0, stream>>>(ew, sbuf, dst, v0w, v1w, agg0, agg1, E);
    node_out_kernel<L><<<nb, 256, 0, stream>>>(agg0, agg1, h0cur, h1cur, wo0, wo1,
                                               s0, b0, s1, b1, h0out, h1out, N);
}

extern "C" void kernel_launch(void* const* d_in, const int* in_sizes, int n_in,
                              void* d_out, int out_size, void* d_ws, size_t ws_size,
                              hipStream_t stream) {
    (void)n_in; (void)out_size; (void)ws_size;
    const float* x     = (const float*)d_in[0];
    const float* v     = (const float*)d_in[1];   // initial h1: (N, 8, 3)
    const int*   src   = (const int*)  d_in[2];
    const int*   dst   = (const int*)  d_in[3];
    const float* efeat = (const float*)d_in[4];
    const int E = in_sizes[2];
    const int N = in_sizes[0] / 3;

    // params flattened per-layer with sorted dict keys
    // L0: b0 b1 rb1 rb2 rw1 rw2 s0 s1 wo0 wo1 wq1            (idx 5..15)
    // L1: b0 b1 rb1 rb2 rw1 rw2 s0 s1 wo0 wo1 wq0 wq1        (idx 16..27)
    // L2: rb1 rb2 rw1 rw2 wo1 wq1                            (idx 28..33)
    const float* L0_b0 = (const float*)d_in[5];  const float* L0_b1 = (const float*)d_in[6];
    const float* L0_rb1= (const float*)d_in[7];  const float* L0_rb2= (const float*)d_in[8];
    const float* L0_rw1= (const float*)d_in[9];  const float* L0_rw2= (const float*)d_in[10];
    const float* L0_s0 = (const float*)d_in[11]; const float* L0_s1 = (const float*)d_in[12];
    const float* L0_wo0= (const float*)d_in[13]; const float* L0_wo1= (const float*)d_in[14];
    const float* L0_wq1= (const float*)d_in[15];
    const float* L1_b0 = (const float*)d_in[16]; const float* L1_b1 = (const float*)d_in[17];
    const float* L1_rb1= (const float*)d_in[18]; const float* L1_rb2= (const float*)d_in[19];
    const float* L1_rw1= (const float*)d_in[20]; const float* L1_rw2= (const float*)d_in[21];
    const float* L1_s0 = (const float*)d_in[22]; const float* L1_s1 = (const float*)d_in[23];
    const float* L1_wo0= (const float*)d_in[24]; const float* L1_wo1= (const float*)d_in[25];
    const float* L1_wq0= (const float*)d_in[26]; const float* L1_wq1= (const float*)d_in[27];
    const float* L2_rb1= (const float*)d_in[28]; const float* L2_rb2= (const float*)d_in[29];
    const float* L2_rw1= (const float*)d_in[30]; const float* L2_rw2= (const float*)d_in[31];
    const float* L2_wo1= (const float*)d_in[32]; const float* L2_wq1= (const float*)d_in[33];

    // workspace carve-up (floats)
    float* ws = (float*)d_ws;
    float* rhat   = ws; ws += (size_t)E*3;
    float* distb  = ws; ws += (size_t)E;
    float* logits = ws; ws += (size_t)E;
    float* ewb    = ws; ws += (size_t)E;
    float* v0w    = ws; ws += (size_t)E*2;
    float* v1w    = ws; ws += (size_t)E*6;
    float* h0A    = ws; ws += (size_t)N*8;
    float* h0B    = ws; ws += (size_t)N*8;
    float* h1A    = ws; ws += (size_t)N*24;
    float* h1B    = ws; ws += (size_t)N*24;
    float* q1b    = ws; ws += (size_t)N*6;
    float* q0b    = ws; ws += (size_t)N*2;
    unsigned* mbuf = (unsigned*)ws; ws += (size_t)N;
    float* sbuf   = ws; ws += (size_t)N;
    float* agg0   = ws; ws += (size_t)N*2;
    float* agg1   = ws; ws += (size_t)N*6;

    int eb = (E + 255) / 256;
    geom_kernel<<<eb, 256, 0, stream>>>(x, src, dst, rhat, distb, E);

    // Layer 0: h1 = v, h0 = none -> h1A, h0A
    run_layer<0>(v, nullptr,
                 L0_rw1, L0_rb1, L0_rw2, L0_rb2, L0_wq1, nullptr, L0_wo1, L0_wo0,
                 L0_s0, L0_b0, L0_s1, L0_b1,
                 h1A, h0A,
                 distb, efeat, rhat, src, dst,
                 q1b, q0b, mbuf, sbuf, agg0, agg1, logits, ewb, v0w, v1w,
                 N, E, stream);
    // Layer 1: h1A,h0A -> h1B,h0B
    run_layer<1>(h1A, h0A,
                 L1_rw1, L1_rb1, L1_rw2, L1_rb2, L1_wq1, L1_wq0, L1_wo1, L1_wo0,
                 L1_s0, L1_b0, L1_s1, L1_b1,
                 h1B, h0B,
                 distb, efeat, rhat, src, dst,
                 q1b, q0b, mbuf, sbuf, agg0, agg1, logits, ewb, v0w, v1w,
                 N, E, stream);
    // Layer 2: h1B,h0B -> d_out (N,2,3)
    run_layer<2>(h1B, h0B,
                 L2_rw1, L2_rb1, L2_rw2, L2_rb2, L2_wq1, nullptr, L2_wo1, nullptr,
                 nullptr, nullptr, nullptr, nullptr,
                 (float*)d_out, nullptr,
                 distb, efeat, rhat, src, dst,
                 q1b, q0b, mbuf, sbuf, agg0, agg1, logits, ewb, v0w, v1w,
                 N, E, stream);
}
